// Model_13048110645270
// MI455X (gfx1250) — compile-verified, matrix operand order checked by reading, not run on hardware
//
#include <hip/hip_runtime.h>
#include <math.h>

#define BSZ 256           // batch
#define NN  1000          // N
// Output layout (flat, return order)
#define X0_OFF    0
#define STEPS_OFF (BSZ*NN)                       // 256000
#define EPS_OFF   (STEPS_OFF + BSZ*(NN-1))       // 511744
#define VAR_OFF   (EPS_OFF + BSZ)                // 512000
#define TS_OFF    (VAR_OFF + BSZ*NN)             // 768000
#define ALPHA_OFF (TS_OFF + BSZ*NN)              // 1024000
#define BETA_OFF  (ALPHA_OFF + BSZ)              // 1024256
#define POWR_OFF  (BETA_OFF + BSZ)               // 1024512

typedef __bf16 v16bf __attribute__((ext_vector_type(16)));
typedef float  v8f   __attribute__((ext_vector_type(8)));

union Frag { unsigned int u[8]; v16bf v; };

__device__ __forceinline__ unsigned int f2bf_u(float f) {
    unsigned int u = __float_as_uint(f);
    return (u + 0x7FFFu + ((u >> 16) & 1u)) >> 16;     // RNE
}
__device__ __forceinline__ unsigned int pack2bf(float lo, float hi) {
    return f2bf_u(lo) | (f2bf_u(hi) << 16);
}

#define A_TILE (64 * 34)
#define B_TILE (128 * 34)

// ---------------------------------------------------------------------------
// bf16 WMMA GEMM, shapes are compile-time constants:
//   C[M,N] = act(A[M,K] @ B[K,N] + bias[N])
// 256 threads (8 wave32), tile 64x128, BK=32, 32x32 per wave,
// double-buffered LDS, packed-b32 staging; all edge handling folds statically.
// ---------------------------------------------------------------------------
template<int M, int N, int K, bool RELU>
__global__ __launch_bounds__(256)
void gemm_bf16_wmma(const float* __restrict__ A, const float* __restrict__ B,
                    const float* __restrict__ bias, float* __restrict__ C) {
    static_assert(M % 64 == 0, "M tiles exactly");
    static_assert(K % 4 == 0 && N % 4 == 0, "16B-aligned rows");
    constexpr int kChunks = (K + 31) >> 5;
    constexpr bool KFULL = (K % 32 == 0);
    constexpr bool NFULL = (N % 128 == 0);

    __shared__ unsigned short As[2 * A_TILE];   // [buf][64][32] pad->34
    __shared__ unsigned short Bs[2 * B_TILE];   // [buf][n][k]   pad->34

    const int tid  = threadIdx.x;
    const int lane = tid & 31;
    const int wave = tid >> 5;
    const int wm   = wave >> 2;          // 0..1
    const int wn   = wave & 3;           // 0..3
    const int half = lane >> 4;
    const int l15  = lane & 15;
    const int m0   = blockIdx.y * 64;
    const int n0   = blockIdx.x * 128;

    // staging assignments
    const int arow = tid >> 3;           // 0..31 -> rows arow, arow+32
    const int acol = (tid & 7) * 4;      // 0..28
    const int bnl  = tid & 127;          // 0..127 (column within tile)
    const int bkp  = tid >> 7;           // 0..1   (k = 2*bkp + 4*s)

    auto stage = [&](int buf, int k0) {
        unsigned short* Asb = As + buf * A_TILE;
        unsigned short* Bsb = Bs + buf * B_TILE;
        // ---- A tile (64x32): only a K-edge possible ----
        if (KFULL || k0 + 32 <= K) {
#pragma unroll
            for (int rr = 0; rr < 2; ++rr) {
                int r = arow + rr * 32;
                const float4 v = *(const float4*)(A + (size_t)(m0 + r) * K + k0 + acol);
                *(unsigned int*)&Asb[r * 34 + acol]     = pack2bf(v.x, v.y);
                *(unsigned int*)&Asb[r * 34 + acol + 2] = pack2bf(v.z, v.w);
            }
        } else {
#pragma unroll
            for (int rr = 0; rr < 2; ++rr) {
                int r = arow + rr * 32;
                float x[4];
#pragma unroll
                for (int c = 0; c < 4; ++c) {
                    int kk = k0 + acol + c;
                    x[c] = (kk < K) ? A[(size_t)(m0 + r) * K + kk] : 0.0f;
                }
                *(unsigned int*)&Asb[r * 34 + acol]     = pack2bf(x[0], x[1]);
                *(unsigned int*)&Asb[r * 34 + acol + 2] = pack2bf(x[2], x[3]);
            }
        }
        // ---- B tile (32x128), transposed into [n][k] ----
        int gn = n0 + bnl;
        if ((KFULL || k0 + 32 <= K) && (NFULL || n0 + 128 <= N)) {
#pragma unroll
            for (int s = 0; s < 8; ++s) {
                int k = 2 * bkp + 4 * s;
                float x0 = B[(size_t)(k0 + k)     * N + gn];
                float x1 = B[(size_t)(k0 + k + 1) * N + gn];
                *(unsigned int*)&Bsb[bnl * 34 + k] = pack2bf(x0, x1);
            }
        } else {
#pragma unroll
            for (int s = 0; s < 8; ++s) {
                int k  = 2 * bkp + 4 * s;
                int g0 = k0 + k, g1 = k0 + k + 1;
                float x0 = ((KFULL || g0 < K) && (NFULL || gn < N)) ? B[(size_t)g0 * N + gn] : 0.0f;
                float x1 = ((KFULL || g1 < K) && (NFULL || gn < N)) ? B[(size_t)g1 * N + gn] : 0.0f;
                *(unsigned int*)&Bsb[bnl * 34 + k] = pack2bf(x0, x1);
            }
        }
    };

    v8f acc[2][2];
#pragma unroll
    for (int i = 0; i < 2; ++i)
#pragma unroll
        for (int j = 0; j < 2; ++j)
#pragma unroll
            for (int r = 0; r < 8; ++r) acc[i][j][r] = 0.0f;

    int kv[8];
#pragma unroll
    for (int v = 0; v < 8; ++v)
        kv[v] = ((v >> 2) << 4) + ((v & 3) << 1) + (half << 3);

    stage(0, 0);
    int cur = 0;
    for (int kc = 0; kc < kChunks; ++kc) {
        __syncthreads();
        const unsigned short* Asb = As + cur * A_TILE;
        const unsigned short* Bsb = Bs + cur * B_TILE;
        // fragment gathers first, so next-chunk staging overlaps the WMMAs
        Frag a[2], b[2];
#pragma unroll
        for (int mt = 0; mt < 2; ++mt) {
            int row = (wm * 32 + mt * 16 + l15) * 34;
#pragma unroll
            for (int v = 0; v < 8; ++v)
                a[mt].u[v] = *(const unsigned int*)&Asb[row + kv[v]];
        }
#pragma unroll
        for (int nt = 0; nt < 2; ++nt) {
            int row = (wn * 32 + nt * 16 + l15) * 34;
#pragma unroll
            for (int v = 0; v < 8; ++v)
                b[nt].u[v] = *(const unsigned int*)&Bsb[row + kv[v]];
        }
        if (kc + 1 < kChunks) stage(cur ^ 1, (kc + 1) << 5);
#pragma unroll
        for (int mt = 0; mt < 2; ++mt)
#pragma unroll
            for (int nt = 0; nt < 2; ++nt)
                acc[mt][nt] = __builtin_amdgcn_wmma_f32_16x16x32_bf16(
                    false, a[mt].v, false, b[nt].v,
                    (short)0, acc[mt][nt], false, false);
        cur ^= 1;
    }

    // epilogue: bias + optional relu (M tiles exactly -> no row guard)
#pragma unroll
    for (int mt = 0; mt < 2; ++mt)
#pragma unroll
        for (int nt = 0; nt < 2; ++nt) {
            int gn = n0 + wn * 32 + nt * 16 + l15;
            if (!NFULL && gn >= N) continue;
            float bv = bias[gn];
#pragma unroll
            for (int r = 0; r < 8; ++r) {
                int gm = m0 + wm * 32 + mt * 16 + r + half * 8;
                float v = acc[mt][nt][r] + bv;
                if (RELU) v = fmaxf(v, 0.0f);
                C[(size_t)gm * N + gn] = v;
            }
        }
}

// ---------------------------------------------------------------------------
// batch-1 MLP layer: y[j] = act(sum_k x[k]*W[k,j] + b[j])
// ---------------------------------------------------------------------------
__global__ void vecmat(const float* __restrict__ x, const float* __restrict__ W,
                       const float* __restrict__ bias, float* __restrict__ y,
                       int K, int Nout, int doRelu) {
    int j = blockIdx.x * blockDim.x + threadIdx.x;
    if (j >= Nout) return;
    float s = bias[j];
    for (int k = 0; k < K; ++k) s = fmaf(x[k], W[(size_t)k * Nout + j], s);
    if (doRelu) s = fmaxf(s, 0.0f);
    y[j] = s;
}

// ---------------------------------------------------------------------------
// rhs / ts / steps / alpha / beta / powr epilogue
// ---------------------------------------------------------------------------
__global__ void postproc(const float* __restrict__ xi, const float* __restrict__ ts,
                         const float* __restrict__ var, float* __restrict__ rhs_ws,
                         float* __restrict__ ts_out, float* __restrict__ steps_out,
                         float* __restrict__ a_out, float* __restrict__ b_out,
                         float* __restrict__ p_out) {
    int b = blockIdx.y;
    int n = blockIdx.x * blockDim.x + threadIdx.x;
    float alpha = xi[0], beta = xi[1];
    float powr  = 1.0f / (1.0f + expf(-xi[2]));
    if (blockIdx.x == 0 && threadIdx.x == 0) {
        a_out[b] = alpha; b_out[b] = beta; p_out[b] = powr;
    }
    if (n >= NN) return;
    float t = ts[n];
    float v = var[b * NN + n];
    float base = fabsf(alpha * t + beta * t * t);
    float r = powf(base, powr) * v * v + t;
    rhs_ws[n * BSZ + b] = r;              // [N][BS] for coalesced solver reads
    ts_out[b * NN + n] = t;
    if (n < NN - 1) steps_out[b * (NN - 1) + n] = 0.001f;  // sigmoid(STEP_LOGIT)
}

// ---------------------------------------------------------------------------
// Block-tridiagonal factorization of AtA + 1e-8 I  (batch-independent, O(N))
// ---------------------------------------------------------------------------
__device__ __forceinline__ void inv3(const float a[9], float o[9]) {
    float c00 = a[4]*a[8] - a[5]*a[7];
    float c01 = a[5]*a[6] - a[3]*a[8];
    float c02 = a[3]*a[7] - a[4]*a[6];
    float id  = 1.0f / (a[0]*c00 + a[1]*c01 + a[2]*c02);
    o[0]=c00*id; o[1]=(a[2]*a[7]-a[1]*a[8])*id; o[2]=(a[1]*a[5]-a[2]*a[4])*id;
    o[3]=c01*id; o[4]=(a[0]*a[8]-a[2]*a[6])*id; o[5]=(a[2]*a[3]-a[0]*a[5])*id;
    o[6]=c02*id; o[7]=(a[1]*a[6]-a[0]*a[7])*id; o[8]=(a[0]*a[4]-a[1]*a[3])*id;
}

__global__ void factor_kernel(float* __restrict__ Uinv, float* __restrict__ Lm) {
    if (threadIdx.x != 0 || blockIdx.x != 0) return;
    const float h = 0.001f, h2 = 0.5f * h * h, eps = 1e-8f;
    const float O[9] = {-1.f, 0.f, 0.f,  -h, -1.f, 0.f,  -h2, -h, 0.f};
    float Up[9];
    for (int j = 0; j < NN; ++j) {
        float D[9] = {0,0,0,0,0,0,0,0,0};
        if (j < NN - 1) {   // own continuity rows (constraint A + B outer products)
            D[0]=1.f;    D[1]=h;        D[2]=h2;
            D[3]=h;      D[4]=h*h;      D[5]=h*h2;
            D[6]=h2;     D[7]=h*h2;     D[8]=h2*h2;
            D[4]+=1.f;   D[5]+=h;       D[7]+=h;    D[8]+=h*h;
        }
        if (j > 0) { D[0] += 1.f; D[4] += 1.f; }   // previous node's rows
        D[4] += 1.f;                               // collocation c=(0,1,0)
        if (j == 0) D[0] += 1.f;                   // initial-value row
        D[0] += eps; D[4] += eps; D[8] += eps;     // +1e-8 I

        float L[9];
        if (j == 0) {
            for (int i = 0; i < 9; ++i) L[i] = 0.f;
        } else {
            for (int r = 0; r < 3; ++r)            // L = O^T * Uinv_{j-1}
                for (int c = 0; c < 3; ++c)
                    L[r*3+c] = O[0*3+r]*Up[0*3+c] + O[1*3+r]*Up[1*3+c] + O[2*3+r]*Up[2*3+c];
            for (int r = 0; r < 3; ++r)            // U = D - L * O
                for (int c = 0; c < 3; ++c)
                    D[r*3+c] -= L[r*3+0]*O[0*3+c] + L[r*3+1]*O[1*3+c] + L[r*3+2]*O[2*3+c];
        }
        inv3(D, Up);
        for (int i = 0; i < 9; ++i) { Uinv[j*9+i] = Up[i]; Lm[j*9+i] = L[i]; }
    }
}

// ---------------------------------------------------------------------------
// Batched block-Thomas solve + residual norm. One lane per batch column.
// ---------------------------------------------------------------------------
__global__ __launch_bounds__(32)
void solve_kernel(const float* __restrict__ Uinv, const float* __restrict__ Lm,
                  const float* __restrict__ rhs_ws, const float* __restrict__ var,
                  float* __restrict__ yws, float* __restrict__ x0_out,
                  float* __restrict__ eps_out) {
    const int b = blockIdx.x * 32 + threadIdx.x;
    const float h = 0.001f, h2 = 0.5f * h * h;
    const float init = var[b * NN];

    // forward: y_j = b_j - L_j y_{j-1}
    float y0 = init, y1 = rhs_ws[b], y2 = 0.f;
    yws[b] = y0; yws[BSZ + b] = y1; yws[2 * BSZ + b] = y2;
    for (int j = 1; j < NN; ++j) {
        const float* L = Lm + j * 9;
        float r1 = rhs_ws[j * BSZ + b];
        float t0 =      -(L[0]*y0 + L[1]*y1 + L[2]*y2);
        float t1 = r1 - (L[3]*y0 + L[4]*y1 + L[5]*y2);
        float t2 =      -(L[6]*y0 + L[7]*y1 + L[8]*y2);
        y0 = t0; y1 = t1; y2 = t2;
        float* yp = yws + (size_t)j * 3 * BSZ + b;
        yp[0] = y0; yp[BSZ] = y1; yp[2 * BSZ] = y2;
    }

    // backward: u_j = Uinv_j (y_j - O u_{j+1}); accumulate residual ssq
    const float* U = Uinv + (NN - 1) * 9;
    float u0 = U[0]*y0 + U[1]*y1 + U[2]*y2;
    float u1 = U[3]*y0 + U[4]*y1 + U[5]*y2;
    float u2 = U[6]*y0 + U[7]*y1 + U[8]*y2;
    float ssq = 0.f;
    { float rc = u1 - rhs_ws[(NN - 1) * BSZ + b]; ssq += rc * rc; }
    x0_out[b * NN + (NN - 1)] = u0;

    for (int j = NN - 2; j >= 0; --j) {
        const float* yp = yws + (size_t)j * 3 * BSZ + b;
        float t0 = yp[0]       + u0;
        float t1 = yp[BSZ]     + h  * u0 + u1;
        float t2 = yp[2 * BSZ] + h2 * u0 + h * u1;
        U = Uinv + j * 9;
        float n0 = U[0]*t0 + U[1]*t1 + U[2]*t2;
        float n1 = U[3]*t0 + U[4]*t1 + U[5]*t2;
        float n2 = U[6]*t0 + U[7]*t1 + U[8]*t2;
        float rA = -n0 - h * n1 - h2 * n2 + u0;     // continuity row A
        float rB = -n1 - h * n2 + u1;               // continuity row B
        float rc =  n1 - rhs_ws[j * BSZ + b];       // collocation row
        ssq += rA * rA + rB * rB + rc * rc;
        x0_out[b * NN + j] = n0;
        u0 = n0; u1 = n1; u2 = n2;
    }
    float r0 = u0 - init;                           // initial-value row
    ssq += r0 * r0;
    eps_out[b] = sqrtf(ssq);
}

// ---------------------------------------------------------------------------
extern "C" void kernel_launch(void* const* d_in, const int* in_sizes, int n_in,
                              void* d_out, int out_size, void* d_ws, size_t ws_size,
                              hipStream_t stream) {
    (void)in_sizes; (void)n_in; (void)out_size; (void)ws_size;
    const float* pin    = (const float*)d_in[0];
    const float* ptime  = (const float*)d_in[1];
    const float* pw1    = (const float*)d_in[2];
    const float* pb1    = (const float*)d_in[3];
    const float* pw2    = (const float*)d_in[4];
    const float* pb2    = (const float*)d_in[5];
    const float* pw3    = (const float*)d_in[6];
    const float* pb3    = (const float*)d_in[7];
    const float* tw1    = (const float*)d_in[8];
    const float* tb1    = (const float*)d_in[9];
    const float* tw2    = (const float*)d_in[10];
    const float* tb2    = (const float*)d_in[11];
    const float* tw3    = (const float*)d_in[12];
    const float* tb3    = (const float*)d_in[13];
    const float* nw1    = (const float*)d_in[14];
    const float* nb1    = (const float*)d_in[15];
    const float* nw2    = (const float*)d_in[16];
    const float* nb2    = (const float*)d_in[17];
    const float* nw3    = (const float*)d_in[18];
    const float* nb3    = (const float*)d_in[19];
    const float* net_iv = (const float*)d_in[20];
    float* out = (float*)d_out;
    float* ws  = (float*)d_ws;

    // workspace layout (floats)
    float* h1p    = ws;                       // 1024
    float* h2p    = ws + 1024;                // 1024
    float* xi     = ws + 2048;                // 3 (pad to 8)
    float* h1t    = ws + 2056;                // 1024
    float* h2t    = ws + 3080;                // 1024
    float* ts     = ws + 4104;                // 1000
    float* H1     = ws + 5104;                // 256*1024
    float* H2     = H1 + 256 * 1024;          // 256*1024
    float* rhs_ws = H2 + 256 * 1024;          // 1000*256
    float* Uinv   = rhs_ws + NN * BSZ;        // 9000
    float* Lm     = Uinv + 9 * NN;            // 9000
    float* yws    = Lm + 9 * NN;              // 1000*3*256

    // batch-independent block-tridiagonal factorization
    factor_kernel<<<1, 32, 0, stream>>>(Uinv, Lm);

    // xi = mlp(pin)
    vecmat<<<4, 256, 0, stream>>>(pin, pw1, pb1, h1p, 64, 1024, 1);
    vecmat<<<4, 256, 0, stream>>>(h1p, pw2, pb2, h2p, 1024, 1024, 1);
    vecmat<<<1, 256, 0, stream>>>(h2p, pw3, pb3, xi, 1024, 3, 0);

    // ts = mlp(ptime)
    vecmat<<<4, 256, 0, stream>>>(ptime, tw1, tb1, h1t, 64, 1024, 1);
    vecmat<<<4, 256, 0, stream>>>(h1t, tw2, tb2, h2t, 1024, 1024, 1);
    vecmat<<<4, 256, 0, stream>>>(h2t, tw3, tb3, ts, 1024, 1000, 0);

    // var = mlp(net_iv) — WMMA GEMM chain, writes var directly into output slot
    gemm_bf16_wmma<256, 1024, 1000, true ><<<dim3(8, 4), 256, 0, stream>>>(net_iv, nw1, nb1, H1);
    gemm_bf16_wmma<256, 1024, 1024, true ><<<dim3(8, 4), 256, 0, stream>>>(H1,     nw2, nb2, H2);
    gemm_bf16_wmma<256, 1000, 1024, false><<<dim3(8, 4), 256, 0, stream>>>(H2,     nw3, nb3, out + VAR_OFF);

    // rhs / ts / steps / scalars
    postproc<<<dim3(4, BSZ), 256, 0, stream>>>(xi, ts, out + VAR_OFF, rhs_ws,
                                               out + TS_OFF, out + STEPS_OFF,
                                               out + ALPHA_OFF, out + BETA_OFF,
                                               out + POWR_OFF);

    // batched banded least-squares solve + residual norms
    solve_kernel<<<8, 32, 0, stream>>>(Uinv, Lm, rhs_ws, out + VAR_OFF, yws,
                                       out + X0_OFF, out + EPS_OFF);
}
